// GConvSeq_7859790152279
// MI455X (gfx1250) — compile-verified, matrix operand order checked by reading, not run on hardware
//
#include <hip/hip_runtime.h>

typedef __attribute__((ext_vector_type(16))) _Float16 v16h;
typedef __attribute__((ext_vector_type(4)))  _Float16 v4h;
typedef __attribute__((ext_vector_type(2)))  _Float16 v2h;
typedef __attribute__((ext_vector_type(8)))  float    v8f;

#define FD 128   // feature dim

// ---------------- degree / norm ----------------

__global__ void deg_init_kernel(float* __restrict__ deg, int n) {
    int i = blockIdx.x * blockDim.x + threadIdx.x;
    if (i < n) deg[i] = 1.0f;                 // self-loop contributes 1
}

__global__ void deg_count_kernel(const int* __restrict__ row, float* __restrict__ deg, int e) {
    int i = blockIdx.x * blockDim.x + threadIdx.x;
    if (i < e) atomicAdd(&deg[row[i]], 1.0f);
}

__global__ void rsqrt_kernel(float* __restrict__ deg, int n) {
    int i = blockIdx.x * blockDim.x + threadIdx.x;
    if (i < n) deg[i] = rsqrtf(deg[i]);
}

// ---------------- utility ----------------

__global__ void zero_kernel(float* __restrict__ p, long n) {
    long i = (long)blockIdx.x * blockDim.x + threadIdx.x;
    if (i < n) p[i] = 0.0f;
}

__global__ void relu_kernel(float* __restrict__ p, long n) {
    long i = (long)blockIdx.x * blockDim.x + threadIdx.x;
    if (i < n) p[i] = fmaxf(p[i], 0.0f);
}

// Convert a weight matrix (FD x FD, f32 row-major) to f16 once per layer.
__global__ void w_to_f16_kernel(const float* __restrict__ W, _Float16* __restrict__ W16) {
    int i = blockIdx.x * blockDim.x + threadIdx.x;   // FD*FD threads
    W16[i] = (_Float16)W[i];
}

// ---------------- GEMM: Out16[M,128] = f16( relu?(A[M,128]) @ W16[128,128]^T + b ) --------
// One wave32 computes a full 16x128 output row-block: A tile converted to f16 once
// (4 x v16h), then 8 N-tiles x 4 k-steps of v_wmma_f32_16x16x32_f16 (32 WMMAs/wave).
// B is pre-converted f16 -> contiguous 16B chunks per half-register (b128 loads).
// Block = 256 threads = 8 waves = 8 M-tiles.

template <int RELU>
__global__ void gemm_bias_wmma_kernel(const float* __restrict__ A,
                                      const _Float16* __restrict__ W16,
                                      const float* __restrict__ bias,
                                      _Float16* __restrict__ Out16,
                                      int M) {
    const int wave   = threadIdx.x >> 5;
    const int lane   = threadIdx.x & 31;
    const int laneHi = lane >> 4;            // 0 or 1
    const int lm     = lane & 15;
    const int mtile  = blockIdx.x * 8 + wave;
    if (mtile * 16 >= M) return;             // wave-uniform exit; live waves keep EXEC all-1s
    const bool fullTile = (mtile * 16 + 16 <= M);               // wave-uniform
    const long arow = (long)min(mtile * 16 + lm, M - 1) * FD;   // clamped A row

    // Preload + convert A (16x128) once.
    // ISA 16-bit A 16x32 layout: VGPR v holds a consecutive K pair; lanes 16-31
    // shift K by +8; VGPRs 4-7 cover K 16..31 of the 32-wide k-step.
    v16h av[4];
    #pragma unroll
    for (int ksi = 0; ksi < 4; ++ksi) {
        #pragma unroll
        for (int v = 0; v < 8; ++v) {
            const int k = ksi * 32 + ((v & 4) << 2) + ((v & 3) << 1) + (laneHi << 3);
            float2 fa = *(const float2*)(A + arow + k);
            if (RELU) { fa.x = fmaxf(fa.x, 0.0f); fa.y = fmaxf(fa.y, 0.0f); }
            av[ksi][2 * v]     = (_Float16)fa.x;
            av[ksi][2 * v + 1] = (_Float16)fa.y;
        }
    }

    #pragma unroll
    for (int nt = 0; nt < 8; ++nt) {
        const int  n    = nt * 16 + lm;      // output column (B/C/D lane role)
        const long brow = (long)n * FD;      // B[k][n] = W[n][k], contiguous in k
        const float bv0 = bias[n];
        v8f acc = {bv0, bv0, bv0, bv0, bv0, bv0, bv0, bv0};

        #pragma unroll
        for (int ksi = 0; ksi < 4; ++ksi) {
            v16h bv;
            #pragma unroll
            for (int v = 0; v < 8; ++v) {
                const int k = ksi * 32 + ((v & 4) << 2) + ((v & 3) << 1) + (laneHi << 3);
                v2h p = *(const v2h*)(W16 + brow + k);   // adjacent v -> 16B contiguous
                bv[2 * v]     = p.x;
                bv[2 * v + 1] = p.y;
            }
            acc = __builtin_amdgcn_wmma_f32_16x16x32_f16(
                /*neg_a=*/false, av[ksi], /*neg_b=*/false, bv,
                /*c_mod=*/(short)0, acc, /*reuse_a=*/false, /*reuse_b=*/false);
        }

        // C/D layout: VGPR r -> row r (lanes 0-15) / r+8 (lanes 16-31)
        const long obase = (long)(mtile * 16 + laneHi * 8) * FD + n;
        if (fullTile) {                       // common path: no per-element exec juggling
            #pragma unroll
            for (int r = 0; r < 8; ++r)
                Out16[obase + (long)r * FD] = (_Float16)acc[r];
        } else {                              // ragged tail tile only
            #pragma unroll
            for (int r = 0; r < 8; ++r) {
                const int m = mtile * 16 + laneHi * 8 + r;
                if (m < M) Out16[(long)m * FD + n] = (_Float16)acc[r];
            }
        }
    }
}

// ---------------- message passing: out[col] += dis[row]*dis[col] * hlin16[row] ------------
// One wave32 per message (E real edges + N self-loops); lane gathers 4 f16 (32*8B = 256B row)
// and accumulates in f32 via global atomics.

__global__ void scatter_kernel(const _Float16* __restrict__ hlin16,
                               const float* __restrict__ nrm,
                               const int* __restrict__ ei,   // [2,E] flat: rows then cols
                               float* __restrict__ out,
                               int E, int N) {
    const int msg  = blockIdx.x * (blockDim.x >> 5) + (threadIdx.x >> 5);
    const int lane = threadIdx.x & 31;
    if (msg >= E + N) return;
    int row, col;
    if (msg < E) { row = ei[msg]; col = ei[E + msg]; }
    else         { row = msg - E; col = row; }            // self-loop
    const float w = nrm[row] * nrm[col];
    const v4h v = *(const v4h*)(hlin16 + (long)row * FD + lane * 4);
    float* dst = out + (long)col * FD + lane * 4;
    atomicAdd(dst + 0, w * (float)v.x);
    atomicAdd(dst + 1, w * (float)v.y);
    atomicAdd(dst + 2, w * (float)v.z);
    atomicAdd(dst + 3, w * (float)v.w);
}

// ---------------- launch ----------------

extern "C" void kernel_launch(void* const* d_in, const int* in_sizes, int n_in,
                              void* d_out, int out_size, void* d_ws, size_t ws_size,
                              hipStream_t stream) {
    const float* x  = (const float*)d_in[0];
    const int*   ei = (const int*)d_in[1];
    const float* W1 = (const float*)d_in[2];
    const float* b1 = (const float*)d_in[3];
    const float* W2 = (const float*)d_in[4];
    const float* b2 = (const float*)d_in[5];
    const float* W3 = (const float*)d_in[6];
    const float* b3 = (const float*)d_in[7];
    float* out = (float*)d_out;

    const int N = in_sizes[0] / FD;
    const int E = in_sizes[1] / 2;
    const long total = (long)N * FD;

    // workspace layout: hlin16 (N*FD f16) | nrm (N f32) | w16 (FD*FD f16)
    _Float16* hlin16 = (_Float16*)d_ws;
    float*    nrm    = (float*)(hlin16 + (size_t)N * FD);
    _Float16* w16    = (_Float16*)(nrm + N);

    const int nBlk   = (N + 255) / 256;
    const int eBlk   = (E + 255) / 256;
    const int tBlk   = (int)((total + 255) / 256);
    const int wBlk   = (FD * FD) / 256;
    const int mTile  = (N + 15) / 16;
    const int gBlk   = (mTile + 7) / 8;          // 8 M-tiles (waves) per block
    const int msgBlk = (E + N + 7) / 8;          // 8 messages (waves) per block

    // normalization: deg over source after self-loops, then rsqrt
    deg_init_kernel<<<nBlk, 256, 0, stream>>>(nrm, N);
    deg_count_kernel<<<eBlk, 256, 0, stream>>>(ei, nrm, E);
    rsqrt_kernel<<<nBlk, 256, 0, stream>>>(nrm, N);

    // layer 1
    w_to_f16_kernel<<<wBlk, 256, 0, stream>>>(W1, w16);
    gemm_bias_wmma_kernel<0><<<gBlk, 256, 0, stream>>>(x, w16, b1, hlin16, N);
    zero_kernel<<<tBlk, 256, 0, stream>>>(out, total);
    scatter_kernel<<<msgBlk, 256, 0, stream>>>(hlin16, nrm, ei, out, E, N);

    // layer 2 (ReLU of layer-1 output fused into A load)
    w_to_f16_kernel<<<wBlk, 256, 0, stream>>>(W2, w16);
    gemm_bias_wmma_kernel<1><<<gBlk, 256, 0, stream>>>(out, w16, b2, hlin16, N);
    zero_kernel<<<tBlk, 256, 0, stream>>>(out, total);
    scatter_kernel<<<msgBlk, 256, 0, stream>>>(hlin16, nrm, ei, out, E, N);

    // layer 3
    w_to_f16_kernel<<<wBlk, 256, 0, stream>>>(W3, w16);
    gemm_bias_wmma_kernel<1><<<gBlk, 256, 0, stream>>>(out, w16, b3, hlin16, N);
    zero_kernel<<<tBlk, 256, 0, stream>>>(out, total);
    scatter_kernel<<<msgBlk, 256, 0, stream>>>(hlin16, nrm, ei, out, E, N);

    // final activation
    relu_kernel<<<tBlk, 256, 0, stream>>>(out, total);
}
